// Attention_Linear_25709674234652
// MI455X (gfx1250) — matrix operand order repaired and checked by reading, not compiled, not run on hardware
//
#include <hip/hip_runtime.h>
#include <hip/hip_bf16.h>

// ---------------------------------------------------------------------------
// Linear attention (phi(u)=sqrt(1+u^2)) on MI455X / gfx1250.
// All GEMMs via v_wmma_f32_16x16x32_bf16 (wave32), f32 accumulation.
// ---------------------------------------------------------------------------

typedef __bf16 bf16_t;
typedef __attribute__((ext_vector_type(16))) __bf16 v16bf;
typedef __attribute__((ext_vector_type(8)))  float  v8f;

static constexpr int Bb = 4;
static constexpr int S  = 8192;
static constexpr int D  = 1024;
static constexpr int H  = 16;
static constexpr int HD = 64;
static constexpr int AD = 64;
static constexpr int M1 = Bb * S;   // 32768
static constexpr int N1 = 3 * D;    // 3072
static constexpr int K1 = D;        // 1024
static constexpr int BH = Bb * H;   // 64
static constexpr float EPS = 1e-6f;

#define WMMA_BF16(a, b, c) \
  __builtin_amdgcn_wmma_f32_16x16x32_bf16(false, (a), false, (b), (short)0, (c), false, false)

// A-operand tile (16x32 bf16, row-major source, leading dim lda).
// ISA layout: lanes 0-15 -> K 0..7 / 16..23 ; lanes 16-31 -> K 8..15 / 24..31.
__device__ __forceinline__ v16bf load_a_tile(const bf16_t* __restrict__ tile, int lda, int lane) {
  const bf16_t* p = tile + (size_t)(lane & 15) * lda + ((lane >> 4) & 1) * 8;
  union { uint4 u[2]; v16bf v; } t;
  t.u[0] = *reinterpret_cast<const uint4*>(p);
  t.u[1] = *reinterpret_cast<const uint4*>(p + 16);
  return t.v;
}

// B-operand tile (32x16 bf16) sourced from a row-major B^T ([N, K], leading dim ldb).
// ISA layout: lanes 0-15 -> K 0..7 / 16..23 ; lanes 16-31 -> K 8..15 / 24..31.
__device__ __forceinline__ v16bf load_bt_tile(const bf16_t* __restrict__ tile, int ldb, int lane) {
  const bf16_t* p = tile + (size_t)(lane & 15) * ldb + ((lane >> 4) & 1) * 8;
  union { uint4 u[2]; v16bf v; } t;
  t.u[0] = *reinterpret_cast<const uint4*>(p);
  t.u[1] = *reinterpret_cast<const uint4*>(p + 16);
  return t.v;
}

// 8 bf16 packed for a single 16-byte store (transposed C stores are s-contiguous)
union Pack8 { bf16_t b[8]; uint4 u; };

// ---------------------------------------------------------------------------
// fp32 -> bf16 conversion (4 elements / thread)
// ---------------------------------------------------------------------------
__global__ void cvt_f32_bf16(const float* __restrict__ in, bf16_t* __restrict__ out, long n) {
  long i4 = ((long)blockIdx.x * blockDim.x + threadIdx.x) * 4;
  if (i4 + 3 < n) {
    float4 f = *reinterpret_cast<const float4*>(in + i4);
    union { bf16_t b[4]; uint2 u; } t;
    t.b[0] = (bf16_t)f.x; t.b[1] = (bf16_t)f.y;
    t.b[2] = (bf16_t)f.z; t.b[3] = (bf16_t)f.w;
    *reinterpret_cast<uint2*>(out + i4) = t.u;
  }
}

__global__ void zero_f32(float* __restrict__ p, int n) {
  int i = blockIdx.x * blockDim.x + threadIdx.x;
  if (i < n) p[i] = 0.0f;
}

// ---------------------------------------------------------------------------
// QKV GEMM: C[M1,N1] = Xbf @ Wbf^T + bias.  Wave tile = 64x64 (16 WMMA / K-step).
// Epilogue scatters into Qh/Kh ([bh, s, hd]) and VT ([bh, hd, s], transposed,
// packed 16B stores).
// ---------------------------------------------------------------------------
__global__ __launch_bounds__(256) void gemm_qkv(
    const bf16_t* __restrict__ X, const bf16_t* __restrict__ W,
    const float* __restrict__ bias,
    bf16_t* __restrict__ Qh, bf16_t* __restrict__ Kh, bf16_t* __restrict__ VT) {
  const int lane = threadIdx.x & 31;
  const int wave = threadIdx.x >> 5;          // 0..7
  const int wm = wave & 3, wn = wave >> 2;    // 4 (m) x 2 (n)
  const int m0 = blockIdx.y * 256 + wm * 64;
  const int n0 = blockIdx.x * 128 + wn * 64;
  const int half = (lane >> 4) & 1;
  const int ln = lane & 15;

  v8f acc[4][4] = {};
  for (int kk = 0; kk < K1; kk += 32) {
    v16bf a[4], b[4];
#pragma unroll
    for (int mi = 0; mi < 4; mi++)
      a[mi] = load_a_tile(X + (size_t)(m0 + 16 * mi) * K1 + kk, K1, lane);
#pragma unroll
    for (int ni = 0; ni < 4; ni++)
      b[ni] = load_bt_tile(W + (size_t)(n0 + 16 * ni) * K1 + kk, K1, lane);
    if (kk + 64 < K1) {
      // pull the streamed X operand into WGP$ two K-steps ahead
#pragma unroll
      for (int mi = 0; mi < 4; mi++)
        __builtin_prefetch(X + (size_t)(m0 + 16 * mi + ln) * K1 + kk + 64 + half * 8, 0, 3);
    }
#pragma unroll
    for (int mi = 0; mi < 4; mi++)
#pragma unroll
      for (int ni = 0; ni < 4; ni++)
        acc[mi][ni] = WMMA_BF16(a[mi], b[ni], acc[mi][ni]);
  }

#pragma unroll
  for (int ni = 0; ni < 4; ni++) {
    const int ng = n0 + 16 * ni + ln;
    const float bn = bias[ng];
    const int which = ng >> 10;      // 0=Q, 1=K, 2=V  (D == 1024)
    const int nl = ng & (D - 1);
    const int h = nl >> 6, hd = nl & 63;
#pragma unroll
    for (int mi = 0; mi < 4; mi++) {
      const int mt = m0 + 16 * mi;
      const int bidx = mt / S;               // uniform over the 16-row tile
      const int sbase = (mt & (S - 1)) + 8 * half;
      const size_t bh = (size_t)(bidx * H + h);
      if (which == 2) {
        Pack8 p;
#pragma unroll
        for (int r = 0; r < 8; r++) p.b[r] = (bf16_t)(acc[mi][ni][r] + bn);
        *reinterpret_cast<uint4*>(&VT[(bh * HD + hd) * S + sbase]) = p.u;
      } else {
        bf16_t* dst = (which == 0) ? Qh : Kh;
#pragma unroll
        for (int r = 0; r < 8; r++)
          dst[(bh * S + sbase + r) * HD + hd] = (bf16_t)(acc[mi][ni][r] + bn);
      }
    }
  }
}

// ---------------------------------------------------------------------------
// p = In @ Wp^T + bp ; phi = sqrt(1 + p*p).
// transposed==0: Out = phiQ [bh, s, a]; transposed==1: Out = phiK^T [bh, a, s].
// ---------------------------------------------------------------------------
__global__ __launch_bounds__(256) void proj_phi(
    const bf16_t* __restrict__ In,   // [BH, S, HD]
    const bf16_t* __restrict__ Wp,   // [AD, HD] row-major (== B^T layout)
    const float* __restrict__ bp,    // [AD]
    bf16_t* __restrict__ Out, int transposed) {
  const int lane = threadIdx.x & 31;
  const int wave = threadIdx.x >> 5;
  const int bh = blockIdx.x;
  const int m0 = blockIdx.y * 256 + wave * 32;
  const bf16_t* A0 = In + (size_t)bh * S * HD;

  v8f acc[2][4] = {};
#pragma unroll
  for (int t = 0; t < 2; t++) {
    const int kk = 32 * t;
    v16bf a[2], b[4];
#pragma unroll
    for (int mi = 0; mi < 2; mi++)
      a[mi] = load_a_tile(A0 + (size_t)(m0 + 16 * mi) * HD + kk, HD, lane);
#pragma unroll
    for (int ni = 0; ni < 4; ni++)
      b[ni] = load_bt_tile(Wp + (size_t)(16 * ni) * HD + kk, HD, lane);
#pragma unroll
    for (int mi = 0; mi < 2; mi++)
#pragma unroll
      for (int ni = 0; ni < 4; ni++)
        acc[mi][ni] = WMMA_BF16(a[mi], b[ni], acc[mi][ni]);
  }

  const int half = (lane >> 4) & 1;
  const int ln = lane & 15;
#pragma unroll
  for (int ni = 0; ni < 4; ni++) {
    const int aidx = 16 * ni + ln;
    const float bn = bp[aidx];
#pragma unroll
    for (int mi = 0; mi < 2; mi++) {
      const int sbase = m0 + 16 * mi + 8 * half;
      if (transposed) {
        Pack8 p;
#pragma unroll
        for (int r = 0; r < 8; r++) {
          const float pv = acc[mi][ni][r] + bn;
          p.b[r] = (bf16_t)sqrtf(1.0f + pv * pv);
        }
        *reinterpret_cast<uint4*>(&Out[((size_t)bh * AD + aidx) * S + sbase]) = p.u;
      } else {
#pragma unroll
        for (int r = 0; r < 8; r++) {
          const float pv = acc[mi][ni][r] + bn;
          Out[((size_t)bh * S + sbase + r) * AD + aidx] = (bf16_t)sqrtf(1.0f + pv * pv);
        }
      }
    }
  }
}

// ---------------------------------------------------------------------------
// KTV[bh] = phiK^T @ V  (64x64 f32, atomics across waves/blocks) and k_sum[bh].
// A = phiK^T [AD, S] row-major; B^T = V^T [HD, S] row-major.
// Grid y splits S into 8 chunks for occupancy/bandwidth.
// ---------------------------------------------------------------------------
__global__ __launch_bounds__(256) void ktv_kernel(
    const bf16_t* __restrict__ PhiKT,  // [BH, AD, S]
    const bf16_t* __restrict__ VT,     // [BH, HD, S]
    float* __restrict__ KTV,           // [BH, AD, HD] (pre-zeroed)
    float* __restrict__ Ksum) {        // [BH, AD]     (pre-zeroed)
  const int lane = threadIdx.x & 31;
  const int wave = threadIdx.x >> 5;
  const int bh = blockIdx.x;
  const int c0 = blockIdx.y * (S / 8);
  const bf16_t* A0 = PhiKT + (size_t)bh * AD * S;
  const bf16_t* B0 = VT + (size_t)bh * HD * S;

  v8f acc[4][4] = {};
  float ks[4] = {0.f, 0.f, 0.f, 0.f};

  for (int kk = c0 + wave * 32; kk < c0 + S / 8; kk += 256) {
    v16bf a[4], b[4];
#pragma unroll
    for (int mi = 0; mi < 4; mi++)
      a[mi] = load_a_tile(A0 + (size_t)(16 * mi) * S + kk, S, lane);
#pragma unroll
    for (int ni = 0; ni < 4; ni++)
      b[ni] = load_bt_tile(B0 + (size_t)(16 * ni) * S + kk, S, lane);
#pragma unroll
    for (int mi = 0; mi < 4; mi++)
#pragma unroll
      for (int ni = 0; ni < 4; ni++)
        acc[mi][ni] = WMMA_BF16(a[mi], b[ni], acc[mi][ni]);
    // k_sum: each lane's A elements are phiK values of row a = 16*mi + (lane&15)
#pragma unroll
    for (int mi = 0; mi < 4; mi++)
#pragma unroll
      for (int e = 0; e < 16; e++)
        ks[mi] += (float)a[mi][e];
  }

  const int half = (lane >> 4) & 1;
  const int ln = lane & 15;
#pragma unroll
  for (int mi = 0; mi < 4; mi++) {
#pragma unroll
    for (int ni = 0; ni < 4; ni++) {
#pragma unroll
      for (int r = 0; r < 8; r++) {
        const int arow = 16 * mi + r + 8 * half;
        const int hcol = 16 * ni + ln;
        atomicAdd(&KTV[((size_t)bh * AD + arow) * HD + hcol], acc[mi][ni][r]);
      }
    }
    atomicAdd(&Ksum[bh * AD + 16 * mi + ln], ks[mi]);
  }
}

// KTV f32 [bh, a, hd] -> bf16 transposed [bh, hd, a] (B^T layout for final GEMM)
__global__ void ktv_cvt(const float* __restrict__ KTV, bf16_t* __restrict__ KTVT) {
  int i = blockIdx.x * blockDim.x + threadIdx.x;
  if (i < BH * AD * HD) {
    int bh = i / (AD * HD);
    int rem = i - bh * (AD * HD);
    int a = rem / HD, hd = rem - a * HD;
    KTVT[((size_t)bh * HD + hd) * AD + a] = (bf16_t)KTV[i];
  }
}

// ---------------------------------------------------------------------------
// out[b, s, h*64+hd] = (phiQ[bh] @ KTV[bh]) / (phiQ[bh] . ksum[bh] + eps)
// Denominator fused: per-lane dot partials from the WMMA A-tiles, combined
// across half-lanes with ds_swizzle (SWAPX16) and broadcast with ds_bpermute.
// ---------------------------------------------------------------------------
__global__ __launch_bounds__(256) void out_kernel(
    const bf16_t* __restrict__ PhiQ,  // [BH, S, AD]
    const bf16_t* __restrict__ KTVT,  // [BH, HD, AD]
    const float* __restrict__ Ksum,   // [BH, AD]
    float* __restrict__ Out) {        // [B, S, D]
  const int lane = threadIdx.x & 31;
  const int wave = threadIdx.x >> 5;
  const int bh = blockIdx.x;
  const int m0 = blockIdx.y * 256 + wave * 32;
  const int b = bh / H, h = bh - b * H;
  const int half = (lane >> 4) & 1;
  const int ln = lane & 15;
  const bf16_t* A0 = PhiQ + (size_t)bh * S * AD;
  const bf16_t* B0 = KTVT + (size_t)bh * HD * AD;
  const float* ks = Ksum + bh * AD;

  // ksum reordered into the A-operand element order for each 32-wide K step:
  // element e at step t has K = 32*t + 16*(e>=8) + 8*half + (e&7)
  float ksv[2][16];
#pragma unroll
  for (int t = 0; t < 2; t++)
#pragma unroll
    for (int e = 0; e < 16; e++)
      ksv[t][e] = ks[32 * t + ((e >> 3) << 4) + half * 8 + (e & 7)];

  v8f acc[2][4] = {};
  float dpart[2] = {0.f, 0.f};
#pragma unroll
  for (int t = 0; t < 2; t++) {
    const int kk = 32 * t;
    v16bf a[2], bm[4];
#pragma unroll
    for (int mi = 0; mi < 2; mi++)
      a[mi] = load_a_tile(A0 + (size_t)(m0 + 16 * mi) * AD + kk, AD, lane);
#pragma unroll
    for (int ni = 0; ni < 4; ni++)
      bm[ni] = load_bt_tile(B0 + (size_t)(16 * ni) * AD + kk, AD, lane);
#pragma unroll
    for (int mi = 0; mi < 2; mi++)
#pragma unroll
      for (int ni = 0; ni < 4; ni++)
        acc[mi][ni] = WMMA_BF16(a[mi], bm[ni], acc[mi][ni]);
    // denominator partials from the same A-tiles
#pragma unroll
    for (int mi = 0; mi < 2; mi++)
#pragma unroll
      for (int e = 0; e < 16; e++)
        dpart[mi] += (float)a[mi][e] * ksv[t][e];
  }

  // lane L now holds a K-half partial of den(row M = L&15); add the other half
  float den[2];
#pragma unroll
  for (int mi = 0; mi < 2; mi++) {
    int o = __builtin_amdgcn_ds_swizzle(__float_as_int(dpart[mi]), 0x401f); // xor 0x10
    den[mi] = dpart[mi] + __int_as_float(o);
  }

#pragma unroll
  for (int mi = 0; mi < 2; mi++) {
#pragma unroll
    for (int r = 0; r < 8; r++) {
      const int s = m0 + 16 * mi + r + 8 * half;
      // broadcast den(row r + 8*half) from the lane that owns it
      const float dv = __int_as_float(
          __builtin_amdgcn_ds_bpermute(4 * (r + 8 * half), __float_as_int(den[mi])));
      const float inv = 1.0f / (dv + EPS);
#pragma unroll
      for (int ni = 0; ni < 4; ni++)
        Out[((size_t)b * S + s) * D + h * HD + 16 * ni + ln] = acc[mi][ni][r] * inv;
    }
  }
}

// ---------------------------------------------------------------------------
extern "C" void kernel_launch(void* const* d_in, const int* in_sizes, int n_in,
                              void* d_out, int out_size, void* d_ws, size_t ws_size,
                              hipStream_t stream) {
  (void)in_sizes; (void)n_in; (void)out_size; (void)ws_size;
  const float* x    = (const float*)d_in[0];
  const float* Wq   = (const float*)d_in[1];
  const float* bq   = (const float*)d_in[2];
  const float* Wp   = (const float*)d_in[3];
  const float* bp   = (const float*)d_in[4];

  char* ws = (char*)d_ws;
  constexpr size_t SZ_X   = (size_t)M1 * K1 * 2;        // 64 MB
  constexpr size_t SZ_W   = (size_t)N1 * K1 * 2;        // 6 MB
  constexpr size_t SZ_WP  = (size_t)AD * HD * 2;        // 8 KB
  constexpr size_t SZ_QKV = (size_t)BH * S * HD * 2;    // 64 MB each

  bf16_t* Xbf   = (bf16_t*)(ws);
  bf16_t* Wqbf  = (bf16_t*)(ws + SZ_X);
  bf16_t* Wpbf  = (bf16_t*)(ws + SZ_X + SZ_W);
  bf16_t* Qh    = (bf16_t*)(ws + SZ_X + SZ_W + SZ_WP);
  bf16_t* Kh    = (bf16_t*)((char*)Qh + SZ_QKV);
  bf16_t* VT    = (bf16_t*)((char*)Kh + SZ_QKV);
  float*  KTV   = (float*)((char*)VT + SZ_QKV);
  float*  Ksum  = KTV + (size_t)BH * AD * HD;
  bf16_t* KTVT  = (bf16_t*)(Ksum + BH * AD);
  // Aliased regions (dead by the time they are reused; stream serializes):
  bf16_t* PhiQ  = Xbf;   // overlays Xbf after gemm_qkv
  bf16_t* PhiKT = Qh;    // overlays Qh after proj_phi(Q)

  // 1. fp32 -> bf16
  cvt_f32_bf16<<<((long)M1 * K1 / 4 + 255) / 256, 256, 0, stream>>>(x, Xbf, (long)M1 * K1);
  cvt_f32_bf16<<<((long)N1 * K1 / 4 + 255) / 256, 256, 0, stream>>>(Wq, Wqbf, (long)N1 * K1);
  cvt_f32_bf16<<<((long)AD * HD / 4 + 255) / 256, 256, 0, stream>>>(Wp, Wpbf, (long)AD * HD);
  // 2. zero KTV + Ksum (contiguous)
  zero_f32<<<((BH * AD * HD + BH * AD) + 255) / 256, 256, 0, stream>>>(KTV, BH * AD * HD + BH * AD);
  // 3. QKV projection GEMM
  gemm_qkv<<<dim3(N1 / 128, M1 / 256), 256, 0, stream>>>(Xbf, Wqbf, bq, Qh, Kh, VT);
  // 4. approx-dim projection + phi (Q then K; order matters for aliasing)
  proj_phi<<<dim3(BH, S / 256), 256, 0, stream>>>(Qh, Wpbf, bp, PhiQ, 0);
  proj_phi<<<dim3(BH, S / 256), 256, 0, stream>>>(Kh, Wpbf, bp, PhiKT, 1);
  // 5. KTV + k_sum reduction over S (S split 8 ways for bandwidth)
  ktv_kernel<<<dim3(BH, 8), 256, 0, stream>>>(PhiKT, VT, KTV, Ksum);
  // 6. KTV -> bf16 (transposed for B-operand)
  ktv_cvt<<<(BH * AD * HD + 255) / 256, 256, 0, stream>>>(KTV, KTVT);
  // 7. numerator GEMM + fused denominator + divide + write output
  out_kernel<<<dim3(BH, S / 256), 256, 0, stream>>>(PhiQ, KTVT, Ksum, (float*)d_out);
}
